// InstanceLoss_10565619548447
// MI455X (gfx1250) — matrix operand order repaired and checked by reading, not compiled, not run on hardware
//
#include <hip/hip_runtime.h>
#include <hip/hip_bf16.h>
#include <stdint.h>

// ---------------- problem constants (match reference) ----------------
#define B_     4
#define E_     32
#define H_     512
#define W_     512
#define NPIX   (H_ * W_)          // 262144
#define KINST  64
#define NSEG   (KINST + 1)        // 65 (label 0 = background)
#define STOT   (B_ * NSEG)        // 260 segments
#define DELTA_V 0.5f
#define DELTA_D 1.5f
#define W_EDGE  10.0f
#define W_NORM  0.001f

// workspace layout (float offsets)
#define OFF_CNT   0                       // [STOT]
#define OFF_WSUM  (OFF_CNT  + STOT)       // [STOT]
#define OFF_WC    (OFF_WSUM + STOT)       // [STOT*E_]
#define OFF_PULL  (OFF_WC   + STOT*E_)    // [STOT]
#define ZERO_LEN  (OFF_PULL + STOT)       // everything up to here must be zeroed
#define OFF_CENT  ZERO_LEN                // [STOT*E_]

typedef __attribute__((ext_vector_type(16))) _Float16 v16h;
typedef __attribute__((ext_vector_type(8)))  float    v8f;

// Optional gfx1250 async global->LDS copy path (ASYNCcnt-tracked).
// Prototype discovered via compiler diagnostic:
//   (int4 __device__* gsrc, int4 __shared__* ldst, imm int, imm int)
#if defined(__has_builtin)
#if __has_builtin(__builtin_amdgcn_global_load_async_to_lds_b128)
#define HAVE_ASYNC_LDS 1
typedef int v4i __attribute__((__vector_size__(16)));
typedef __attribute__((address_space(1))) v4i as1_v4i;   // global (printed as __device__)
typedef __attribute__((address_space(3))) v4i as3_v4i;   // LDS (printed as __shared__)
#endif
#endif
#ifndef HAVE_ASYNC_LDS
#define HAVE_ASYNC_LDS 0
#endif

// ---------------- k0: zero accumulators ----------------
__global__ void k_init(float* __restrict__ ws, int n) {
    for (int i = blockIdx.x * blockDim.x + threadIdx.x; i < n;
         i += gridDim.x * blockDim.x)
        ws[i] = 0.0f;
}

// Shared 3x6 label window for 4 consecutive pixels in a row.
// Returns bitmask of non-background pixels; fills segment ids and edge weights.
// Boundary (3x3 replicate-pad max != min) <=> not all 9 labels equal.
__device__ __forceinline__ int label_quad(const int* __restrict__ lb, int y, int x0,
                                          int bbase, int sj[4], float wj[4]) {
    int Lv[3][6];
#pragma unroll
    for (int r = 0; r < 3; ++r) {
        int yy = y - 1 + r; yy = yy < 0 ? 0 : (yy > H_ - 1 ? H_ - 1 : yy);
#pragma unroll
        for (int c = 0; c < 6; ++c) {
            int xx = x0 - 1 + c; xx = xx < 0 ? 0 : (xx > W_ - 1 ? W_ - 1 : xx);
            Lv[r][c] = lb[yy * W_ + xx];
        }
    }
    int mask = 0;
#pragma unroll
    for (int j = 0; j < 4; ++j) {
        int c = Lv[1][j + 1];
        bool same = true;
#pragma unroll
        for (int r = 0; r < 3; ++r)
#pragma unroll
            for (int cc = 0; cc < 3; ++cc) same = same && (Lv[r][j + cc] == c);
        wj[j] = same ? 1.0f : W_EDGE;
        sj[j] = bbase + c;
        if (c != 0) mask |= (1 << j);
    }
    return mask;
}

// ---------------- k1: segmented accumulation (cnt, wsum, weighted center sums)
__global__ void __launch_bounds__(256)
k_acc(const float* __restrict__ embed, const int* __restrict__ label,
      float* __restrict__ ws) {
    __shared__ float cnt_sh[STOT];
    __shared__ float wsum_sh[STOT];
    __shared__ float wc_sh[STOT * E_];   // ~33 KB

    for (int i = threadIdx.x; i < STOT; i += blockDim.x) { cnt_sh[i] = 0.f; wsum_sh[i] = 0.f; }
    for (int i = threadIdx.x; i < STOT * E_; i += blockDim.x) wc_sh[i] = 0.f;
    __syncthreads();

    const int total4 = B_ * NPIX / 4;
    const int stride = gridDim.x * blockDim.x;
    for (int idx = blockIdx.x * blockDim.x + threadIdx.x; idx < total4; idx += stride) {
        int p = idx * 4;                       // 4 consecutive pixels, same row
        int b = p / NPIX;
        int n = p - b * NPIX;
        int y = n >> 9, x0 = n & (W_ - 1);
        int sj[4]; float wj[4];
        int mask = label_quad(label + b * NPIX, y, x0, b * NSEG, sj, wj);
#pragma unroll
        for (int j = 0; j < 4; ++j) {
            if ((mask >> j) & 1) {
                atomicAdd(&cnt_sh[sj[j]], 1.0f);
                atomicAdd(&wsum_sh[sj[j]], wj[j]);
            }
        }
        if (!mask) continue;
        const float4* eb = (const float4*)(embed + (size_t)b * E_ * NPIX + p);
        if (idx + stride < total4)
            __builtin_prefetch((const char*)eb + (size_t)stride * 16, 0, 0);
#pragma unroll
        for (int e = 0; e < E_; ++e) {
            float4 v = eb[(size_t)e * (NPIX / 4)];
            float vv[4] = {v.x, v.y, v.z, v.w};
#pragma unroll
            for (int j = 0; j < 4; ++j)
                if ((mask >> j) & 1)
                    atomicAdd(&wc_sh[sj[j] * E_ + e], vv[j] * wj[j]);
        }
    }
    __syncthreads();

    float* cnt_g  = ws + OFF_CNT;
    float* wsum_g = ws + OFF_WSUM;
    float* wc_g   = ws + OFF_WC;
    for (int i = threadIdx.x; i < STOT; i += blockDim.x) {
        if (cnt_sh[i]  != 0.f) atomicAdd(&cnt_g[i],  cnt_sh[i]);
        if (wsum_sh[i] != 0.f) atomicAdd(&wsum_g[i], wsum_sh[i]);
    }
    for (int i = threadIdx.x; i < STOT * E_; i += blockDim.x)
        if (wc_sh[i] != 0.f) atomicAdd(&wc_g[i], wc_sh[i]);
}

// ---------------- k2: centers = wc / (wsum + 1e-8) ----------------
__global__ void k_centers(float* __restrict__ ws) {
    int i = blockIdx.x * blockDim.x + threadIdx.x;
    if (i >= STOT * E_) return;
    int s = i / E_;
    ws[OFF_CENT + i] = ws[OFF_WC + i] / (ws[OFF_WSUM + s] + 1e-8f);
}

// ---------------- k3: pull term (per-pixel distance to own center) ----------------
__global__ void __launch_bounds__(256)
k_pull(const float* __restrict__ embed, const int* __restrict__ label,
       float* __restrict__ ws) {
    __shared__ float cent_sh[STOT * E_];  // ~33 KB
    __shared__ float pull_sh[STOT];

#if HAVE_ASYNC_LDS
    // async copy engine: global -> LDS, tracked by ASYNCcnt
    for (int i = threadIdx.x * 4; i < STOT * E_; i += blockDim.x * 4)
        __builtin_amdgcn_global_load_async_to_lds_b128(
            (as1_v4i*)(uintptr_t)(ws + OFF_CENT + i),
            (as3_v4i*)(unsigned)(uintptr_t)(cent_sh + i), 0, 0);
#else
    for (int i = threadIdx.x; i < STOT * E_; i += blockDim.x) cent_sh[i] = ws[OFF_CENT + i];
#endif
    for (int i = threadIdx.x; i < STOT; i += blockDim.x) pull_sh[i] = 0.f;
#if HAVE_ASYNC_LDS
#if __has_builtin(__builtin_amdgcn_s_wait_asynccnt)
    __builtin_amdgcn_s_wait_asynccnt(0);
#else
    asm volatile("s_wait_asynccnt 0x0" ::: "memory");
#endif
#endif
    __syncthreads();

    const int total4 = B_ * NPIX / 4;
    const int stride = gridDim.x * blockDim.x;
    for (int idx = blockIdx.x * blockDim.x + threadIdx.x; idx < total4; idx += stride) {
        int p = idx * 4;
        int b = p / NPIX;
        int n = p - b * NPIX;
        int y = n >> 9, x0 = n & (W_ - 1);
        int sj[4]; float wj[4];
        int mask = label_quad(label + b * NPIX, y, x0, b * NSEG, sj, wj);
        if (!mask) continue;                   // bg pull_sum is sliced away in reference
        const float4* eb = (const float4*)(embed + (size_t)b * E_ * NPIX + p);
        if (idx + stride < total4)
            __builtin_prefetch((const char*)eb + (size_t)stride * 16, 0, 0);
        float d2[4] = {0.f, 0.f, 0.f, 0.f};
#pragma unroll
        for (int e = 0; e < E_; ++e) {
            float4 v = eb[(size_t)e * (NPIX / 4)];
            float vv[4] = {v.x, v.y, v.z, v.w};
#pragma unroll
            for (int j = 0; j < 4; ++j) {
                float d = vv[j] - cent_sh[sj[j] * E_ + e];
                d2[j] = fmaf(d, d, d2[j]);
            }
        }
#pragma unroll
        for (int j = 0; j < 4; ++j) {
            if ((mask >> j) & 1) {
                float dist = sqrtf(fmaxf(d2[j], 1e-12f));
                float r = fmaxf(dist - DELTA_V, 0.f);
                atomicAdd(&pull_sh[sj[j]], r * r * wj[j]);
            }
        }
    }
    __syncthreads();
    float* pull_g = ws + OFF_PULL;
    for (int i = threadIdx.x; i < STOT; i += blockDim.x)
        if (pull_sh[i] != 0.f) atomicAdd(&pull_g[i], pull_sh[i]);
}

// ---------------- k4: WMMA Gram matrix -> push + norm + final combine ----------------
// 1 block, 4 waves; wave b handles batch b. Gram G = C*C^T via
// v_wmma_f32_16x16x32_f16 (K = 32 = embedding dim), diagonals kept exact f32.
__global__ void __launch_bounds__(128)
k_final(const float* __restrict__ ws, float* __restrict__ out) {
    __shared__ float diag_sh[B_][KINST];
    __shared__ float pres_sh[B_][KINST];
    __shared__ float red_sh[B_][4];      // b_pull, b_push, b_norm, valid

    const int lane = threadIdx.x & 31;
    const int b    = threadIdx.x >> 5;
    const float* cnt  = ws + OFF_CNT;
    const float* pull = ws + OFF_PULL;
    const float* cent = ws + OFF_CENT;

    // exact diagonal |c_i|^2 and presence flags
    for (int i = lane; i < KINST; i += 32) {
        const float* c = cent + (size_t)(b * NSEG + 1 + i) * E_;
        float ss = 0.f;
#pragma unroll
        for (int e = 0; e < E_; ++e) ss = fmaf(c[e], c[e], ss);
        diag_sh[b][i] = ss;
        pres_sh[b][i] = (cnt[b * NSEG + 1 + i] > 0.f) ? 1.f : 0.f;
    }
    __syncthreads();

    // per-wave pull / norm / instance-count partials
    float n_inst_f = 0.f, pull_acc = 0.f, norm_acc = 0.f;
    for (int i = lane; i < KINST; i += 32) {
        if (pres_sh[b][i] != 0.f) {
            n_inst_f += 1.f;
            float c_ = cnt[b * NSEG + 1 + i];
            pull_acc += pull[b * NSEG + 1 + i] / fmaxf(c_, 1.f);
            norm_acc += sqrtf(diag_sh[b][i]);
        }
    }
#pragma unroll
    for (int off = 16; off; off >>= 1) {
        n_inst_f += __shfl_xor(n_inst_f, off, 32);
        pull_acc += __shfl_xor(pull_acc, off, 32);
        norm_acc += __shfl_xor(norm_acc, off, 32);
    }

    // Gram matrix via WMMA; ISA 7.12.2 wave32 fragment layouts.
    const int half = lane >> 4;          // 0: lanes 0-15, 1: lanes 16-31
    const int lm   = lane & 15;
    float push_acc = 0.f;
#pragma unroll
    for (int it = 0; it < 4; ++it) {
        // A fragment: 16x32 f16, row M = it*16 + lm
        v16h a;
        const float* arow = cent + (size_t)(b * NSEG + 1 + it * 16 + lm) * E_;
#pragma unroll
        for (int i2 = 0; i2 < 16; ++i2) {
            int vg = i2 >> 1, grp = vg >> 2;
            int k = grp * 16 + half * 8 + (vg & 3) * 2 + (i2 & 1);
            a[i2] = (_Float16)arow[k];
        }
#pragma unroll
        for (int jt = 0; jt < 4; ++jt) {
            // B fragment: 32x16 f16, B[k][n] = C[jt*16+n][k] (C^T), col N = lm
            v16h bb;
            const float* brow = cent + (size_t)(b * NSEG + 1 + jt * 16 + lm) * E_;
#pragma unroll
            for (int i2 = 0; i2 < 16; ++i2)
                bb[i2] = (_Float16)brow[(i2 >> 3) * 16 + half * 8 + (i2 & 7)];

            v8f cacc = {};
            cacc = __builtin_amdgcn_wmma_f32_16x16x32_f16(
                false, a, false, bb, (short)0, cacc, false, false);

            // D layout: VGPR r -> M = it*16 + half*8 + r, N = jt*16 + lm
#pragma unroll
            for (int r = 0; r < 8; ++r) {
                int gi = it * 16 + half * 8 + r;
                int gj = jt * 16 + lm;
                if (gj > gi && pres_sh[b][gi] != 0.f && pres_sh[b][gj] != 0.f) {
                    float d2 = diag_sh[b][gi] + diag_sh[b][gj] - 2.f * cacc[r];
                    float pw = sqrtf(fmaxf(d2, 0.f));
                    float rr = fmaxf(2.f * DELTA_D - pw, 0.f);
                    push_acc += rr * rr;
                }
            }
        }
    }
#pragma unroll
    for (int off = 16; off; off >>= 1) push_acc += __shfl_xor(push_acc, off, 32);

    if (lane == 0) {
        float n_pairs = n_inst_f * (n_inst_f - 1.f) * 0.5f;
        red_sh[b][0] = pull_acc / fmaxf(n_inst_f, 1.f);
        red_sh[b][1] = push_acc / fmaxf(n_pairs, 1.f);
        red_sh[b][2] = norm_acc / fmaxf(n_inst_f, 1.f);
        red_sh[b][3] = (n_inst_f > 0.f) ? 1.f : 0.f;
    }
    __syncthreads();
    if (threadIdx.x == 0) {
        float sp = 0.f, su = 0.f, sn = 0.f, v = 0.f;
#pragma unroll
        for (int i = 0; i < B_; ++i) {
            sp += red_sh[i][0]; su += red_sh[i][1];
            sn += red_sh[i][2]; v  += red_sh[i][3];
        }
        out[0] = (sp + su + W_NORM * sn) / fmaxf(v, 1.f);
    }
}

// ---------------- host launcher ----------------
extern "C" void kernel_launch(void* const* d_in, const int* in_sizes, int n_in,
                              void* d_out, int out_size, void* d_ws, size_t ws_size,
                              hipStream_t stream) {
    (void)in_sizes; (void)n_in; (void)out_size; (void)ws_size;
    const float* embed = (const float*)d_in[0];
    const int*   label = (const int*)d_in[1];
    float* ws  = (float*)d_ws;
    float* out = (float*)d_out;

    k_init<<<(ZERO_LEN + 255) / 256, 256, 0, stream>>>(ws, ZERO_LEN);
    k_acc<<<512, 256, 0, stream>>>(embed, label, ws);
    k_centers<<<(STOT * E_ + 255) / 256, 256, 0, stream>>>(ws);
    k_pull<<<512, 256, 0, stream>>>(embed, label, ws);
    k_final<<<1, 128, 0, stream>>>(ws, out);
}
